// GAT_27925877358998
// MI455X (gfx1250) — matrix operand order repaired and checked by reading, not compiled, not run on hardware
//
#include <hip/hip_runtime.h>
#include <hip/hip_bf16.h>

#define N_NODES 50000
#define N_EDGES 600000
#define DIM 128
#define HEADS 4
#define HD 512          // HEADS*DIM
#define NLAYERS 6

typedef __attribute__((ext_vector_type(16))) __bf16 v16bf;
typedef __attribute__((ext_vector_type(8)))  __bf16 v8bf;
typedef __attribute__((ext_vector_type(8)))  float  v8f;

__device__ __forceinline__ unsigned short f2bf(float f){
  unsigned u = __float_as_uint(f);
  u += 0x7fffu + ((u >> 16) & 1u);        // round-to-nearest-even
  return (unsigned short)(u >> 16);
}
__device__ __forceinline__ float lrelu(float z){ return z > 0.f ? z : 0.2f * z; }
__device__ __forceinline__ float eluf(float z){ return z > 0.f ? z : (__expf(z) - 1.f); }

__device__ __forceinline__ void atomicMaxF(float* addr, float v){
  if (v >= 0.f) atomicMax((int*)addr, __float_as_int(v));
  else          atomicMin((unsigned int*)addr, __float_as_uint(v));
}

// ---------------- weight conversion: f32 [L,128,512] -> bf16 transposed [mat][n=512][k=128]
__global__ __launch_bounds__(256) void k_conv_w(const float* __restrict__ Ws,
                                                const float* __restrict__ rWs,
                                                unsigned short* __restrict__ wt){
  int gid = blockIdx.x * 256 + threadIdx.x;
  if (gid >= NLAYERS * 2 * HD * DIM) return;
  int mat = gid >> 16;          // HD*DIM = 65536
  int r   = gid & 65535;
  int n   = r >> 7;             // 0..511
  int k   = r & 127;            // 0..127
  int l   = mat >> 1;
  const float* srcp = (mat & 1) ? rWs : Ws;
  wt[gid] = f2bf(srcp[(size_t)l * (DIM * HD) + (size_t)k * HD + n]);
}

// ---------------- x conversion: f32 -> bf16
__global__ __launch_bounds__(256) void k_conv_x(const float* __restrict__ x,
                                                unsigned short* __restrict__ xb){
  int gid = blockIdx.x * 256 + threadIdx.x;
  if (gid >= N_NODES * DIM) return;
  xb[gid] = f2bf(x[gid]);
}

// ---------------- WMMA GEMM: C[M,512] = A[M,128](bf16) x Bt[512,128]^T(bf16), f32 acc
__global__ __launch_bounds__(256) void k_gemm(const unsigned short* __restrict__ Abf,
                                              const unsigned short* __restrict__ Btbf,
                                              float* __restrict__ C, int M){
  const __bf16* A  = (const __bf16*)Abf;
  const __bf16* Bt = (const __bf16*)Btbf;
  const int tid  = threadIdx.x;
  const int wid  = tid >> 5;
  const int lane = tid & 31;
  const int kh   = lane >> 4;        // half-wave selector
  const int ln   = lane & 15;
  const int waveM = wid >> 1;        // 0..3
  const int waveN = wid & 1;         // 0..1
  const int row0 = blockIdx.x * 128 + waveM * 32;    // wave: 32 rows
  const int col0 = blockIdx.y * 128 + waveN * 64;    // wave: 64 cols

  v8f acc[2][4] = {};

  const __bf16* arow[2];
#pragma unroll
  for (int mi = 0; mi < 2; ++mi){
    int r = row0 + mi * 16 + ln;
    if (r >= M) r = M - 1;           // clamp; store is guarded
    arow[mi] = A + (size_t)r * DIM;
  }
  const __bf16* brow[4];
#pragma unroll
  for (int ni = 0; ni < 4; ++ni)
    brow[ni] = Bt + (size_t)(col0 + ni * 16 + ln) * DIM;

#pragma unroll
  for (int kb = 0; kb < DIM; kb += 32){
    v16bf af[2];
#pragma unroll
    for (int mi = 0; mi < 2; ++mi){
      // A 16x32 bf16, contiguous k order used here (same as B): lanes 0-15 K {0..15}, lanes 16-31 K {16..31}
      union { v16bf v; v8bf p[2]; } u;
      u.p[0] = *(const v8bf*)(arow[mi] + kb + kh * 16);
      u.p[1] = *(const v8bf*)(arow[mi] + kb + kh * 16 + 8);
      af[mi] = u.v;
    }
    v16bf bfr[4];
#pragma unroll
    for (int ni = 0; ni < 4; ++ni)
      // B 32x16 bf16, same contiguous k order as A: lane = column, 16 consecutive K per half-wave
      bfr[ni] = *(const v16bf*)(brow[ni] + kb + kh * 16);

#pragma unroll
    for (int mi = 0; mi < 2; ++mi)
#pragma unroll
      for (int ni = 0; ni < 4; ++ni)
        acc[mi][ni] = __builtin_amdgcn_wmma_f32_16x16x32_bf16(
            false, af[mi], false, bfr[ni], (short)0, acc[mi][ni], false, false);
  }

  // C/D layout: lane n = col, VGPR r -> row r (+8 for upper half-wave)
#pragma unroll
  for (int mi = 0; mi < 2; ++mi){
    int mbase = row0 + mi * 16 + kh * 8;
#pragma unroll
    for (int r = 0; r < 8; ++r){
      int m = mbase + r;
      if (m < M){
        float* cp = C + (size_t)m * HD + col0 + ln;
#pragma unroll
        for (int ni = 0; ni < 4; ++ni)
          cp[ni * 16] = acc[mi][ni][r];
      }
    }
  }
}

// ---------------- per-node attention dots: el/er [N,4]
__global__ __launch_bounds__(256) void k_el_er(const float* __restrict__ hbuf,
                                               const float* __restrict__ al,
                                               const float* __restrict__ ar,
                                               float* __restrict__ el,
                                               float* __restrict__ er){
  int wid = threadIdx.x >> 5, lane = threadIdx.x & 31;
  int node = blockIdx.x * 8 + wid;
  if (node >= N_NODES) return;
  const float4* hp = (const float4*)(hbuf + (size_t)node * HD + lane * 16);
  const float4* ap = (const float4*)(al + lane * 16);
  const float4* bp = (const float4*)(ar + lane * 16);
  float sl = 0.f, sr = 0.f;
#pragma unroll
  for (int j = 0; j < 4; ++j){
    float4 hv = hp[j], av = ap[j], bv = bp[j];
    sl += hv.x*av.x + hv.y*av.y + hv.z*av.z + hv.w*av.w;
    sr += hv.x*bv.x + hv.y*bv.y + hv.z*bv.z + hv.w*bv.w;
  }
#pragma unroll
  for (int m = 1; m < 8; m <<= 1){      // reduce 8-lane head segments
    sl += __shfl_xor(sl, m, 32);
    sr += __shfl_xor(sr, m, 32);
  }
  if ((lane & 7) == 0){
    int hh = lane >> 3;
    el[node * HEADS + hh] = sl;
    er[node * HEADS + hh] = sr;
  }
}

// ---------------- init softmax state (and graph-pool accumulator)
__global__ __launch_bounds__(256) void k_init(float* __restrict__ emax,
                                              float* __restrict__ denom,
                                              float* __restrict__ s_acc){
  int gid = blockIdx.x * 256 + threadIdx.x;
  if (gid < N_NODES * HEADS){
    emax[gid]  = -3.0e38f;
    denom[gid] = 0.f;
  }
  if (gid < DIM) s_acc[gid] = 0.f;
}

// ---------------- edge pass 1: segment max
__global__ __launch_bounds__(256) void k_edge_max(const int* __restrict__ src,
                                                  const int* __restrict__ dst,
                                                  const float* __restrict__ el,
                                                  const float* __restrict__ er,
                                                  float* __restrict__ emax){
  int gid = blockIdx.x * 256 + threadIdx.x;
  if (gid >= N_EDGES * HEADS) return;
  int e = gid >> 2, hh = gid & 3;
  int s = src[e], d = dst[e];
  float z = lrelu(el[s * HEADS + hh] + er[d * HEADS + hh]);
  atomicMaxF(&emax[d * HEADS + hh], z);
}

// ---------------- edge pass 2: segment sum of exp
__global__ __launch_bounds__(256) void k_edge_sum(const int* __restrict__ src,
                                                  const int* __restrict__ dst,
                                                  const float* __restrict__ el,
                                                  const float* __restrict__ er,
                                                  const float* __restrict__ emax,
                                                  float* __restrict__ denom){
  int gid = blockIdx.x * 256 + threadIdx.x;
  if (gid >= N_EDGES * HEADS) return;
  int e = gid >> 2, hh = gid & 3;
  int s = src[e], d = dst[e];
  float z = lrelu(el[s * HEADS + hh] + er[d * HEADS + hh]);
  atomicAdd(&denom[d * HEADS + hh], __expf(z - emax[d * HEADS + hh]));
}

// ---------------- edge pass 3: weighted scatter-add of h[src] into rst[dst]
__global__ __launch_bounds__(256) void k_edge_aggr(const int* __restrict__ src,
                                                   const int* __restrict__ dst,
                                                   const float* __restrict__ el,
                                                   const float* __restrict__ er,
                                                   const float* __restrict__ emax,
                                                   const float* __restrict__ denom,
                                                   const float* __restrict__ hbuf,
                                                   float* __restrict__ rst){
  int gid = blockIdx.x * 256 + threadIdx.x;
  if (gid >= N_EDGES * 128) return;       // 128 threads per edge (4 dims each)
  int e  = gid >> 7;
  int t  = gid & 127;
  int hh = t >> 5;                        // head (whole wave shares edge+head)
  int l4 = t & 31;
  int s = src[e], d = dst[e];
  float z = lrelu(el[s * HEADS + hh] + er[d * HEADS + hh]);
  float a = __expf(z - emax[d * HEADS + hh]) / denom[d * HEADS + hh];
  const float4 hv = *(const float4*)(hbuf + (size_t)s * HD + hh * DIM + l4 * 4);
  float* rp = rst + (size_t)d * HD + hh * DIM + l4 * 4;
  atomicAdd(rp + 0, hv.x * a);
  atomicAdd(rp + 1, hv.y * a);
  atomicAdd(rp + 2, hv.z * a);
  atomicAdd(rp + 3, hv.w * a);
}

// ---------------- bias + ELU + head mean + LayerNorm (+ graph-pool accumulation)
__global__ __launch_bounds__(256) void k_finalize(const float* __restrict__ rst,
                                                  const float* __restrict__ bias,
                                                  const float* __restrict__ gamma,
                                                  const float* __restrict__ beta,
                                                  float* __restrict__ xout,
                                                  float* __restrict__ s_acc,
                                                  int is_last){
  int wid = threadIdx.x >> 5, lane = threadIdx.x & 31;
  int node = blockIdx.x * 8 + wid;
  if (node >= N_NODES) return;
  const float* rp = rst + (size_t)node * HD;
  int d0 = lane * 4;
  float y0 = 0.f, y1 = 0.f, y2 = 0.f, y3 = 0.f;
#pragma unroll
  for (int hh = 0; hh < HEADS; ++hh){
    float4 v = *(const float4*)(rp + hh * DIM + d0);
    float4 b = *(const float4*)(bias + hh * DIM + d0);
    y0 += eluf(v.x + b.x);
    y1 += eluf(v.y + b.y);
    y2 += eluf(v.z + b.z);
    y3 += eluf(v.w + b.w);
  }
  y0 *= 0.25f; y1 *= 0.25f; y2 *= 0.25f; y3 *= 0.25f;
  float s  = y0 + y1 + y2 + y3;
  float sq = y0*y0 + y1*y1 + y2*y2 + y3*y3;
#pragma unroll
  for (int m = 1; m < 32; m <<= 1){
    s  += __shfl_xor(s,  m, 32);
    sq += __shfl_xor(sq, m, 32);
  }
  float mu  = s * (1.f / 128.f);
  float var = sq * (1.f / 128.f) - mu * mu;
  float inv = rsqrtf(var + 1e-5f);
  float4 g  = *(const float4*)(gamma + d0);
  float4 be = *(const float4*)(beta + d0);
  float o0 = g.x * (y0 - mu) * inv + be.x;
  float o1 = g.y * (y1 - mu) * inv + be.y;
  float o2 = g.z * (y2 - mu) * inv + be.z;
  float o3 = g.w * (y3 - mu) * inv + be.w;
  *(float4*)(xout + (size_t)node * DIM + d0) = make_float4(o0, o1, o2, o3);
  if (is_last){
    atomicAdd(&s_acc[d0 + 0], o0);
    atomicAdd(&s_acc[d0 + 1], o1);
    atomicAdd(&s_acc[d0 + 2], o2);
    atomicAdd(&s_acc[d0 + 3], o3);
  }
}

__global__ void k_s_final(const float* __restrict__ s_acc, float* __restrict__ out){
  int t = threadIdx.x;
  if (t < DIM) out[t] = s_acc[t] * (1.0f / (float)N_NODES);
}

extern "C" void kernel_launch(void* const* d_in, const int* in_sizes, int n_in,
                              void* d_out, int out_size, void* d_ws, size_t ws_size,
                              hipStream_t stream){
  const float* feats  = (const float*)d_in[0];
  const int*   srcI   = (const int*)d_in[1];
  const int*   dstI   = (const int*)d_in[2];
  const float* Ws     = (const float*)d_in[3];
  const float* attnL  = (const float*)d_in[4];
  const float* attnR  = (const float*)d_in[5];
  const float* biases = (const float*)d_in[6];
  const float* rWs    = (const float*)d_in[7];
  const float* gamma  = (const float*)d_in[8];
  const float* beta   = (const float*)d_in[9];
  float* out = (float*)d_out;

  char* p = (char*)d_ws;
  float* hbuf = (float*)p;                  p += (size_t)N_NODES * HD * 4;   // 102.4 MB
  float* rst  = (float*)p;                  p += (size_t)N_NODES * HD * 4;   // 102.4 MB
  float* xbuf = (float*)p;                  p += (size_t)N_NODES * DIM * 4;  // 25.6 MB
  unsigned short* xb = (unsigned short*)p;  p += (size_t)N_NODES * DIM * 2;  // 12.8 MB
  unsigned short* wt = (unsigned short*)p;  p += (size_t)NLAYERS * 2 * HD * DIM * 2; // 1.5 MB
  float* el    = (float*)p;                 p += (size_t)N_NODES * HEADS * 4;
  float* er    = (float*)p;                 p += (size_t)N_NODES * HEADS * 4;
  float* emax  = (float*)p;                 p += (size_t)N_NODES * HEADS * 4;
  float* denom = (float*)p;                 p += (size_t)N_NODES * HEADS * 4;
  float* s_acc = (float*)p;                 p += 512;

  k_conv_w<<<(NLAYERS * 2 * HD * DIM) / 256, 256, 0, stream>>>(Ws, rWs, wt);

  dim3 ggrid((N_NODES + 127) / 128, HD / 128);
  for (int l = 0; l < NLAYERS; ++l){
    const float* xin = (l == 0) ? feats : xbuf;
    k_conv_x<<<(N_NODES * DIM) / 256, 256, 0, stream>>>(xin, xb);
    k_gemm<<<ggrid, 256, 0, stream>>>(xb, wt + (size_t)(2 * l)     * HD * DIM, hbuf, N_NODES);
    k_gemm<<<ggrid, 256, 0, stream>>>(xb, wt + (size_t)(2 * l + 1) * HD * DIM, rst,  N_NODES);
    k_el_er<<<N_NODES / 8, 256, 0, stream>>>(hbuf, attnL + l * HD, attnR + l * HD, el, er);
    k_init<<<(N_NODES * HEADS + 255) / 256, 256, 0, stream>>>(emax, denom, s_acc);
    k_edge_max<<<(N_EDGES * HEADS) / 256, 256, 0, stream>>>(srcI, dstI, el, er, emax);
    k_edge_sum<<<(N_EDGES * HEADS) / 256, 256, 0, stream>>>(srcI, dstI, el, er, emax, denom);
    k_edge_aggr<<<(N_EDGES * 128) / 256, 256, 0, stream>>>(srcI, dstI, el, er, emax, denom, hbuf, rst);
    k_finalize<<<N_NODES / 8, 256, 0, stream>>>(rst, biases + l * HD, gamma, beta,
                                                (l == NLAYERS - 1) ? out : xbuf,
                                                s_acc, (l == NLAYERS - 1) ? 1 : 0);
  }
  k_s_final<<<1, 128, 0, stream>>>(s_acc, out + (size_t)N_NODES * DIM);
}
